// Bank_38671885534126
// MI455X (gfx1250) — matrix operand order repaired and checked by reading, not compiled, not run on hardware
//
#include <hip/hip_runtime.h>
#include <hip/hip_bf16.h>
#include <stdint.h>

#define NQ 65536
#define DQ 256
#define DM 512
#define NT 128
#define NH 8
#define DH 64

typedef __bf16 bf16_t;
typedef __bf16 v16bf __attribute__((ext_vector_type(16)));
typedef float  v8f   __attribute__((ext_vector_type(8)));

union FragU  { uint32_t u[8]; v16bf v; };
union PackBF { bf16_t h[2]; uint32_t u; };

static __device__ inline uint32_t pack_bf16(float a, float b) {
  PackBF p; p.h[0] = (bf16_t)a; p.h[1] = (bf16_t)b; return p.u;
}

// ---------------- prep kernels (tiny: 128x512 bank/K/V, 256x512 Wq) ----------------

__global__ __launch_bounds__(256) void tanh_kernel(const float* __restrict__ e,
                                                   float* __restrict__ bank) {
  int i = blockIdx.x * 256 + threadIdx.x;
  if (i < NT * DM) bank[i] = tanhf(e[i]);
}

// K = tanh(embed)@Wk as bf16 row-major [t, c]; V transposed to [c, t] for B-fragments
__global__ __launch_bounds__(256) void kv_kernel(const float* __restrict__ bank,
                                                 const float* __restrict__ Wk,
                                                 const float* __restrict__ Wv,
                                                 bf16_t* __restrict__ Kbf,
                                                 bf16_t* __restrict__ Vt) {
  int i = blockIdx.x * 256 + threadIdx.x;
  if (i >= NT * DM) return;
  int t = i / DM, c = i % DM;
  const float* br = bank + t * DM;
  float ak = 0.f, av = 0.f;
  for (int j = 0; j < DM; ++j) {
    float b = br[j];
    ak += b * Wk[j * DM + c];
    av += b * Wv[j * DM + c];
  }
  Kbf[t * DM + c] = (bf16_t)ak;
  Vt[c * NT + t]  = (bf16_t)av;
}

// WqT[n, k] = Wq[k, n] as bf16 so B-fragment K-pairs are contiguous 32-bit loads
__global__ __launch_bounds__(256) void wq_kernel(const float* __restrict__ Wq,
                                                 bf16_t* __restrict__ WqT) {
  int i = blockIdx.x * 256 + threadIdx.x;   // i = k*512 + n (coalesced read)
  if (i >= DM * DQ) return;
  int k = i >> 9;
  int n = i & (DM - 1);
  WqT[n * DQ + k] = (bf16_t)Wq[i];
}

// ---------------- fused attention kernel ----------------
// block = 16 query rows, wave w (of 8) = head w

__global__ __launch_bounds__(256) void attn_kernel(
    const float*  __restrict__ z,      // [NQ, DQ] f32
    const bf16_t* __restrict__ WqT,    // [DM, DQ]
    const bf16_t* __restrict__ Kbf,    // [NT, DM]
    const bf16_t* __restrict__ Vt,     // [DM, NT]
    float* __restrict__ out,           // [NQ, DM]
    float* __restrict__ scores) {      // [NQ, NH*NT]
  __shared__ __align__(16) uint32_t ztp[16 * DQ / 2];   // z tile, packed bf16 pairs (8 KB)
  __shared__ __align__(16) bf16_t   qs[NH][16 * DH];    // per-head q, bf16 (16 KB)
  __shared__ __align__(16) bf16_t   ps[NH][16 * NT];    // per-head probs, bf16 (32 KB)

  const int tid  = threadIdx.x;
  const int h    = tid >> 5;          // wave index == head
  const int lane = tid & 31;
  const int lo   = lane & 15;
  const int hi   = lane >> 4;
  const int row0 = blockIdx.x << 4;

  const v8f zero8 = {0.f, 0.f, 0.f, 0.f, 0.f, 0.f, 0.f, 0.f};

  // ---- stage z tile (16x256 f32 -> packed bf16) ----
  {
    const float* zp = z + (size_t)row0 * DQ + tid * 16;
    #pragma unroll
    for (int i = 0; i < 8; ++i)
      ztp[tid * 8 + i] = pack_bf16(zp[2 * i], zp[2 * i + 1]);
  }
  __syncthreads();

  // ---- q-projection: q[16,64] = ztile[16,256] @ Wq[:, head] (8 k-steps x 4 n-tiles) ----
  v8f accq[4];
  #pragma unroll
  for (int j = 0; j < 4; ++j) accq[j] = zero8;
  #pragma unroll
  for (int s = 0; s < 8; ++s) {
    FragU a;
    #pragma unroll
    for (int i = 0; i < 8; ++i) {
      int kk = s * 32 + ((i & 4) << 2) + hi * 8 + (i & 3) * 2;
      a.u[i] = ztp[lo * (DQ / 2) + (kk >> 1)];
    }
    #pragma unroll
    for (int j = 0; j < 4; ++j) {
      const uint32_t* wp = (const uint32_t*)(WqT + (size_t)(h * DH + j * 16 + lo) * DQ);
      FragU b;
      #pragma unroll
      for (int i = 0; i < 8; ++i) b.u[i] = wp[s * 16 + ((i & 4) << 1) + hi * 4 + (i & 3)];
      accq[j] = __builtin_amdgcn_wmma_f32_16x16x32_bf16(
          false, a.v, false, b.v, (short)0, accq[j], false, false);
    }
  }

  // q * (1/sqrt(64)) -> bf16 LDS (exact power-of-two scale folded in)
  #pragma unroll
  for (int j = 0; j < 4; ++j)
    #pragma unroll
    for (int v = 0; v < 8; ++v)
      qs[h][(v + hi * 8) * DH + j * 16 + lo] = (bf16_t)(accq[j][v] * 0.125f);
  __syncthreads();

  // ---- scores: s[16,128] = q[16,64] @ K^T (2 k-steps x 8 t-tiles) ----
  v8f accs[8];
  #pragma unroll
  for (int j = 0; j < 8; ++j) accs[j] = zero8;
  const uint32_t* qp = (const uint32_t*)&qs[h][0];
  #pragma unroll
  for (int s = 0; s < 2; ++s) {
    FragU a;
    #pragma unroll
    for (int i = 0; i < 8; ++i) {
      int kk = s * 32 + ((i & 4) << 2) + hi * 8 + (i & 3) * 2;
      a.u[i] = qp[lo * (DH / 2) + (kk >> 1)];
    }
    #pragma unroll
    for (int jt = 0; jt < 8; ++jt) {
      const uint32_t* kp = (const uint32_t*)(Kbf + (size_t)(jt * 16 + lo) * DM + h * DH);
      FragU b;
      #pragma unroll
      for (int i = 0; i < 8; ++i) b.u[i] = kp[s * 16 + ((i & 4) << 1) + hi * 4 + (i & 3)];
      accs[jt] = __builtin_amdgcn_wmma_f32_16x16x32_bf16(
          false, a.v, false, b.v, (short)0, accs[jt], false, false);
    }
  }

  // ---- softmax over T in registers (row M = v + 8*hi spans 16 lanes of same hi) ----
  float mrow[8], irow[8];
  #pragma unroll
  for (int v = 0; v < 8; ++v) {
    float m = accs[0][v];
    #pragma unroll
    for (int jt = 1; jt < 8; ++jt) m = fmaxf(m, accs[jt][v]);
    #pragma unroll
    for (int d = 1; d < 16; d <<= 1) m = fmaxf(m, __shfl_xor(m, d, 32));
    mrow[v] = m;
  }
  #pragma unroll
  for (int v = 0; v < 8; ++v) {
    float sum = 0.f;
    #pragma unroll
    for (int jt = 0; jt < 8; ++jt) {
      float e = __expf(accs[jt][v] - mrow[v]);
      accs[jt][v] = e;
      sum += e;
    }
    #pragma unroll
    for (int d = 1; d < 16; d <<= 1) sum += __shfl_xor(sum, d, 32);
    irow[v] = 1.f / sum;
  }

  // ---- write probs: f32 to global scores (coalesced over lo) + bf16 to LDS ----
  {
    float* sc = scores + (size_t)row0 * (NH * NT) + (size_t)h * NT;
    #pragma unroll
    for (int jt = 0; jt < 8; ++jt)
      #pragma unroll
      for (int v = 0; v < 8; ++v) {
        float p = accs[jt][v] * irow[v];
        int M = v + hi * 8;
        sc[(size_t)M * (NH * NT) + jt * 16 + lo] = p;
        ps[h][M * NT + jt * 16 + lo] = (bf16_t)p;
      }
  }
  __syncthreads();

  // ---- out: o[16,64] = P[16,128] @ V[128,64] (4 k-steps x 4 n-tiles) ----
  v8f acco[4];
  #pragma unroll
  for (int j = 0; j < 4; ++j) acco[j] = zero8;
  const uint32_t* pp = (const uint32_t*)&ps[h][0];
  #pragma unroll
  for (int s = 0; s < 4; ++s) {
    FragU a;
    #pragma unroll
    for (int i = 0; i < 8; ++i) {
      int kk = s * 32 + ((i & 4) << 2) + hi * 8 + (i & 3) * 2;
      a.u[i] = pp[lo * (NT / 2) + (kk >> 1)];
    }
    #pragma unroll
    for (int j = 0; j < 4; ++j) {
      const uint32_t* vp = (const uint32_t*)(Vt + (size_t)(h * DH + j * 16 + lo) * NT);
      FragU b;
      #pragma unroll
      for (int i = 0; i < 8; ++i) b.u[i] = vp[s * 16 + ((i & 4) << 1) + hi * 4 + (i & 3)];
      acco[j] = __builtin_amdgcn_wmma_f32_16x16x32_bf16(
          false, a.v, false, b.v, (short)0, acco[j], false, false);
    }
  }

  float* op = out + (size_t)row0 * DM + h * DH;
  #pragma unroll
  for (int j = 0; j < 4; ++j)
    #pragma unroll
    for (int v = 0; v < 8; ++v)
      op[(size_t)(v + hi * 8) * DM + j * 16 + lo] = acco[j][v];
}

// ---------------- launcher ----------------

extern "C" void kernel_launch(void* const* d_in, const int* in_sizes, int n_in,
                              void* d_out, int out_size, void* d_ws, size_t ws_size,
                              hipStream_t stream) {
  (void)in_sizes; (void)n_in; (void)out_size; (void)ws_size;
  const float* z     = (const float*)d_in[0];   // [NQ, DQ]
  const float* embed = (const float*)d_in[1];   // [NT, DM]
  const float* Wq    = (const float*)d_in[2];   // [DQ, DM]
  const float* Wk    = (const float*)d_in[3];   // [DM, DM]
  const float* Wv    = (const float*)d_in[4];   // [DM, DM]

  float* out    = (float*)d_out;                // [NQ, DM]
  float* scores = out + (size_t)NQ * DM;        // [NQ, NH*NT]

  char* ws = (char*)d_ws;
  float*  bank = (float*)ws;                                    // 256 KB
  bf16_t* WqT  = (bf16_t*)(ws + (256 << 10));                   // 256 KB
  bf16_t* Kbf  = (bf16_t*)(ws + (512 << 10));                   // 128 KB
  bf16_t* Vt   = (bf16_t*)(ws + (640 << 10));                   // 128 KB

  tanh_kernel<<<(NT * DM + 255) / 256, 256, 0, stream>>>(embed, bank);
  kv_kernel  <<<(NT * DM + 255) / 256, 256, 0, stream>>>(bank, Wk, Wv, Kbf, Vt);
  wq_kernel  <<<(DM * DQ + 255) / 256, 256, 0, stream>>>(Wq, WqT);
  attn_kernel<<<NQ / 16, 256, 0, stream>>>(z, WqT, Kbf, Vt, out, scores);
}
